// JukeBoxVectorQuantizer_28887950033220
// MI455X (gfx1250) — matrix operand order repaired and checked by reading, not compiled, not run on hardware
//
#include <hip/hip_runtime.h>
#include <hip/hip_bf16.h>

typedef __attribute__((ext_vector_type(16))) _Float16 v16h;
typedef __attribute__((ext_vector_type(8)))  _Float16 v8h;
typedef __attribute__((ext_vector_type(4)))  _Float16 v4h;
typedef __attribute__((ext_vector_type(8)))  float    v8f;

#define NTOK 8192
#define DM   1024
#define DC   64
#define KC   8192

__device__ __forceinline__ v16h cat8(v8h lo, v8h hi) {
  return __builtin_shufflevector(lo, hi, 0,1,2,3,4,5,6,7,8,9,10,11,12,13,14,15);
}

__device__ __forceinline__ v8f wmma16(v16h a, v16h b, v8f c) {
  // D = A(16x32 f16) * B(32x16 f16) + C(16x16 f32)
  return __builtin_amdgcn_wmma_f32_16x16x32_f16(false, a, false, b, (short)0, c, false, false);
}

// fast tanh: v_exp_f32 + v_rcp_f32 (no IEEE-divide fixup chain); >> f16 accuracy
__device__ __forceinline__ float fast_tanh(float x) {
  float xc = fminf(fmaxf(x, -15.0f), 15.0f);
  float e = __expf(2.0f * xc);
  return (e - 1.0f) * __builtin_amdgcn_rcpf(e + 1.0f);
}

// A fragment (16x32 f16) from row-major tile (stride S halves).
// Lane L: m=L&15, kh=(L>>4)*8 ; v[0..7]=A[m][kh..kh+7], v[8..15]=A[m][kh+16..kh+23]
__device__ __forceinline__ v16h load_a_frag(const _Float16* tile, int S, int lane) {
  int m  = lane & 15;
  int kh = (lane >> 4) * 8;
  const _Float16* p = tile + m * S + kh;
  return cat8(*(const v8h*)(p), *(const v8h*)(p + 16));
}

// B fragment (32x16 f16) from K-transposed tile bt[n][k] (stride S halves).
// Lane L: n=L&15, k0=(L>>4)*8 ; v[0..7]=bt[n][k0..k0+7], v[8..15]=bt[n][k0+16..k0+23]
__device__ __forceinline__ v16h load_b_frag(const _Float16* bt, int S, int lane) {
  int n  = lane & 15;
  int k0 = (lane >> 4) * 8;
  const _Float16* p = bt + n * S + k0;
  return cat8(*(const v8h*)(p), *(const v8h*)(p + 16));
}

// ---------------- prep kernels ----------------
__global__ void k_prep_codex(const float* __restrict__ cx, _Float16* __restrict__ ch,
                             float* __restrict__ cn2) {
  int r = blockIdx.x * blockDim.x + threadIdx.x;
  if (r < KC) {
    float s = 0.f;
    for (int d = 0; d < DC; ++d) {
      float v = cx[(size_t)r * DC + d];
      s += v * v;
      ch[(size_t)r * DC + d] = (_Float16)v;
    }
    cn2[r] = s;
  }
}

__global__ void k_twp(const float* __restrict__ Wp, _Float16* __restrict__ WpT) {
  int i = blockIdx.x * blockDim.x + threadIdx.x;  // 64*1024
  if (i < DC * DM) {
    int k = i >> 10, n = i & (DM - 1);
    WpT[(size_t)n * DC + k] = (_Float16)Wp[i];
  }
}

// zero the 256MB one-hot region; base is only 4B-aligned in d_out, so head=3, tail=1
__global__ void k_zero_probs(float* __restrict__ probs) {
  long long tid = (long long)blockIdx.x * blockDim.x + threadIdx.x;
  long long nthreads = (long long)gridDim.x * blockDim.x;
  if (tid == 0) {
    probs[0] = 0.f; probs[1] = 0.f; probs[2] = 0.f;
    probs[(long long)NTOK * KC - 1] = 0.f;
  }
  float4* p4 = (float4*)(probs + 3);                      // 16B aligned
  const long long n4 = ((long long)NTOK * KC - 4) / 4;    // 16777215
  float4 z; z.x = 0.f; z.y = 0.f; z.z = 0.f; z.w = 0.f;
  for (long long i = tid; i < n4; i += nthreads) p4[i] = z;
}

// ---------------- GEMM1: H16 = tanh(Z @ W1 + b1), M=8192 N=1024 K=1024 ----------------
__global__ __launch_bounds__(256) void k_gemm1(const float* __restrict__ Z,
                                               const float* __restrict__ W1,
                                               const float* __restrict__ b1,
                                               _Float16* __restrict__ H16) {
  __shared__ __align__(16) _Float16 As[128 * 32];  // row-major [m][k]
  __shared__ __align__(16) _Float16 Bs[64 * 32];   // transposed [n][k]
  __shared__ __align__(16) _Float16 Os[128 * 64];  // staged output tile
  int t = threadIdx.x, lane = t & 31, w = t >> 5;
  int m0 = blockIdx.y * 128;
  int n0 = blockIdx.x * 64;
  int mw = (w >> 1) * 32, nw = (w & 1) * 32;

  v8f acc[2][2] = {};
  for (int k0 = 0; k0 < DM; k0 += 32) {
    // stage A: 128x32 f32 -> f16 ; float4 global loads, b64 LDS stores
    for (int i = 0; i < 4; ++i) {
      int ch = t + i * 256;                       // 1024 chunks of 4
      int r = ch >> 3, c4 = (ch & 7) * 4;
      const float* gp = &Z[(size_t)(m0 + r) * DM + k0 + c4];
      __builtin_prefetch(gp + 32, 0, 1);          // next k-tile -> global_prefetch
      float4 v = *(const float4*)gp;
      v4h h; h[0] = (_Float16)v.x; h[1] = (_Float16)v.y;
             h[2] = (_Float16)v.z; h[3] = (_Float16)v.w;
      *(v4h*)&As[r * 32 + c4] = h;
    }
    // stage B transposed: W1[k][n] -> Bs[n][k] ; float4 global loads
    for (int i = 0; i < 2; ++i) {
      int ch = t + i * 256;                       // 512 chunks of 4
      int kk = ch >> 4, n4 = (ch & 15) * 4;
      float4 v = *(const float4*)&W1[(size_t)(k0 + kk) * DM + n0 + n4];
      Bs[(n4 + 0) * 32 + kk] = (_Float16)v.x;
      Bs[(n4 + 1) * 32 + kk] = (_Float16)v.y;
      Bs[(n4 + 2) * 32 + kk] = (_Float16)v.z;
      Bs[(n4 + 3) * 32 + kk] = (_Float16)v.w;
    }
    __syncthreads();
    v16h af[2], bf[2];
    af[0] = load_a_frag(&As[(mw) * 32], 32, lane);
    af[1] = load_a_frag(&As[(mw + 16) * 32], 32, lane);
    bf[0] = load_b_frag(&Bs[(nw) * 32], 32, lane);
    bf[1] = load_b_frag(&Bs[(nw + 16) * 32], 32, lane);
    for (int mi = 0; mi < 2; ++mi)
      for (int ni = 0; ni < 2; ++ni)
        acc[mi][ni] = wmma16(af[mi], bf[ni], acc[mi][ni]);
    __syncthreads();
  }
  // epilogue: tanh+convert into LDS tile, then coalesced b128 stores
  int nl = lane & 15, mh = (lane >> 4) * 8;
  for (int mi = 0; mi < 2; ++mi)
    for (int ni = 0; ni < 2; ++ni) {
      int n = nw + ni * 16 + nl;
      float bb = b1[n0 + n];
      for (int i = 0; i < 8; ++i)
        Os[(mw + mi * 16 + mh + i) * 64 + n] = (_Float16)fast_tanh(acc[mi][ni][i] + bb);
    }
  __syncthreads();
  for (int i = 0; i < 4; ++i) {
    int ch = t + i * 256;                         // 1024 chunks of 8 halves
    int r = ch >> 3, c8 = (ch & 7) * 8;
    *(v8h*)&H16[(size_t)(m0 + r) * DM + n0 + c8] = *(const v8h*)&Os[r * 64 + c8];
  }
}

// ---------------- GEMM2: E = H16 @ W2 + b2 ; Z_e = l2norm(E), M=8192 N=64 K=1024 -----
__global__ __launch_bounds__(256) void k_gemm2(const _Float16* __restrict__ H16,
                                               const float* __restrict__ W2,
                                               const float* __restrict__ b2,
                                               float* __restrict__ Ze,
                                               _Float16* __restrict__ Ze16) {
  __shared__ __align__(16) _Float16 As[128 * 32];
  __shared__ __align__(16) _Float16 Bs[64 * 32];  // transposed [n][k]
  __shared__ float Es[128 * 64];
  int t = threadIdx.x, lane = t & 31, w = t >> 5;
  int m0 = blockIdx.x * 128;
  v8f acc[4] = {};
  for (int k0 = 0; k0 < DM; k0 += 32) {
    // stage A via CDNA5 async global->LDS DMA (no VGPR round-trip, ASYNCcnt-tracked)
    for (int i = 0; i < 2; ++i) {
      int ch = t + i * 256;                       // 512 chunks of 8 halves (16B)
      int r = ch >> 2, c8 = (ch & 3) * 8;
      unsigned lds = (unsigned)(unsigned long long)(&As[r * 32 + c8]);
      const _Float16* g = &H16[(size_t)(m0 + r) * DM + k0 + c8];
      asm volatile("global_load_async_to_lds_b128 %0, %1, off"
                   :: "v"(lds), "v"(g) : "memory");
    }
    for (int i = 0; i < 2; ++i) {
      int ch = t + i * 256;                       // 512 chunks of 4 floats
      int kk = ch >> 4, n4 = (ch & 15) * 4;
      float4 v = *(const float4*)&W2[(size_t)(k0 + kk) * DC + n4];
      Bs[(n4 + 0) * 32 + kk] = (_Float16)v.x;
      Bs[(n4 + 1) * 32 + kk] = (_Float16)v.y;
      Bs[(n4 + 2) * 32 + kk] = (_Float16)v.z;
      Bs[(n4 + 3) * 32 + kk] = (_Float16)v.w;
    }
    asm volatile("s_wait_asynccnt 0" ::: "memory");
    __syncthreads();
    v16h a = load_a_frag(&As[(w * 16) * 32], 32, lane);
    for (int ni = 0; ni < 4; ++ni) {
      v16h b = load_b_frag(&Bs[(ni * 16) * 32], 32, lane);
      acc[ni] = wmma16(a, b, acc[ni]);
    }
    __syncthreads();
  }
  int nl = lane & 15, mh = (lane >> 4) * 8;
  for (int ni = 0; ni < 4; ++ni) {
    int n = ni * 16 + nl;
    float bb = b2[n];
    for (int i = 0; i < 8; ++i) Es[(w * 16 + mh + i) * DC + n] = acc[ni][i] + bb;
  }
  __syncthreads();
  if (t < 128) {
    float ss = 0.f;
    for (int d = 0; d < DC; ++d) { float v = Es[t * DC + d]; ss += v * v; }
    float sc = 1.0f / fmaxf(sqrtf(ss), 1e-12f);
    size_t row = (size_t)(m0 + t) * DC;
    for (int d = 0; d < DC; ++d) {
      float v = Es[t * DC + d] * sc;
      Ze[row + d] = v;
      Ze16[row + d] = (_Float16)v;
    }
  }
}

// ---------------- argmax over codes: argmin dist == argmax (2*z.c - |c|^2) ------------
__global__ __launch_bounds__(256) void k_argmax(const _Float16* __restrict__ Ze16,
                                                const _Float16* __restrict__ Ch,
                                                const float* __restrict__ cn2,
                                                int* __restrict__ idxo,
                                                float* __restrict__ probs) {
  __shared__ float bvs[8 * 32 * 8];
  __shared__ int   bis[8 * 32 * 8];
  int t = threadIdx.x, lane = t & 31, w = t >> 5;
  int m0 = blockIdx.x * 128 + w * 16;  // this wave's 16 tokens

  // A fragments (K=64 -> two 16x32 frags), resident for the whole code scan
  v16h a0, a1;
  {
    int m = lane & 15, kh = (lane >> 4) * 8;
    const _Float16* p = Ze16 + (size_t)(m0 + m) * DC + kh;
    a0 = cat8(*(const v8h*)(p),      *(const v8h*)(p + 16));
    a1 = cat8(*(const v8h*)(p + 32), *(const v8h*)(p + 48));
  }
  float bestv[8];
  int   besti[8];
  for (int i = 0; i < 8; ++i) { bestv[i] = -3.0e38f; besti[i] = 0; }

  int nl = lane & 15, k0f = (lane >> 4) * 8;
  for (int c0 = 0; c0 < KC; c0 += 16) {
    int code = c0 + nl;
    const _Float16* bp = Ch + (size_t)code * DC + k0f;
    v16h b0 = cat8(*(const v8h*)(bp),      *(const v8h*)(bp + 16));
    v16h b1 = cat8(*(const v8h*)(bp + 32), *(const v8h*)(bp + 48));
    v8f acc = {};
    acc = wmma16(a0, b0, acc);
    acc = wmma16(a1, b1, acc);
    float c2 = cn2[code];
    for (int i = 0; i < 8; ++i) {
      float s = 2.0f * acc[i] - c2;
      if (s > bestv[i]) { bestv[i] = s; besti[i] = code; }  // ascending scan => first-max kept
    }
  }
  for (int i = 0; i < 8; ++i) {
    bvs[(w * 32 + lane) * 8 + i] = bestv[i];
    bis[(w * 32 + lane) * 8 + i] = besti[i];
  }
  __syncthreads();
  if (t < 128) {
    int ww = t >> 4, ml = t & 15;
    int half = ml >> 3, ii = ml & 7;
    float bb = -3.0e38f;
    int bidx = 0x7fffffff;
    for (int n = 0; n < 16; ++n) {
      int L = half * 16 + n;
      float v = bvs[(ww * 32 + L) * 8 + ii];
      int id = bis[(ww * 32 + L) * 8 + ii];
      if (v > bb || (v == bb && id < bidx)) { bb = v; bidx = id; }
    }
    int token = blockIdx.x * 128 + ww * 16 + ml;
    idxo[token] = bidx;
    probs[(size_t)token * KC + bidx] = 1.0f;  // probs region pre-zeroed earlier on stream
  }
}

// ---------------- VQ gather + commitment loss partials + LayerNorm -> qln16 ----------
__global__ __launch_bounds__(256) void k_vq_ln(const float* __restrict__ Ze,
                                               const float* __restrict__ codex,
                                               const int* __restrict__ idxo,
                                               const float* __restrict__ lnw,
                                               const float* __restrict__ lnb,
                                               _Float16* __restrict__ qln16,
                                               float* __restrict__ partial) {
  __shared__ float red[256];
  int t = threadIdx.x;
  int token = blockIdx.x * 256 + t;
  const float* zrow = Ze + (size_t)token * DC;
  const float* crow = codex + (size_t)idxo[token] * DC;
  float x[DC];
  float lsum = 0.f, mu = 0.f;
  for (int d = 0; d < DC; ++d) {
    float c = crow[d];
    x[d] = c;
    mu += c;
    float df = c - zrow[d];
    lsum += df * df;
  }
  mu *= (1.0f / DC);
  float var = 0.f;
  for (int d = 0; d < DC; ++d) { float dd = x[d] - mu; var += dd * dd; }
  var *= (1.0f / DC);
  float rs = rsqrtf(var + 1e-5f);
  for (int d = 0; d < DC; ++d)
    qln16[(size_t)token * DC + d] = (_Float16)((x[d] - mu) * rs * lnw[d] + lnb[d]);
  red[t] = lsum;
  __syncthreads();
  for (int s = 128; s > 0; s >>= 1) { if (t < s) red[t] += red[t + s]; __syncthreads(); }
  if (t == 0) partial[blockIdx.x] = red[0];
}

__global__ void k_loss_final(const float* __restrict__ partial, float* __restrict__ out_loss) {
  if (threadIdx.x == 0) {
    float s = 0.f;
    for (int i = 0; i < 32; ++i) s += partial[i];  // fixed order -> deterministic
    *out_loss = s * (1.0f / ((float)NTOK * (float)DC));  // BETA = 1.0
  }
}

// ---------------- GEMM3: out = qln16 @ Wp + bp, M=8192 N=1024 K=64 -------------------
__global__ __launch_bounds__(256) void k_gemm3(const _Float16* __restrict__ qln16,
                                               const _Float16* __restrict__ WpT,  // [n][k]
                                               const float* __restrict__ bp,
                                               float* __restrict__ out) {
  __shared__ __align__(16) float Os[128 * 64];  // staged output tile (32KB)
  int t = threadIdx.x, lane = t & 31, w = t >> 5;
  int m0 = blockIdx.y * 128, n0 = blockIdx.x * 64;
  int mw = (w >> 1) * 32, nw = (w & 1) * 32;
  v8f acc[2][2] = {};

  int ml = lane & 15, kh = (lane >> 4) * 8;
  v16h af[2][2];
  for (int mi = 0; mi < 2; ++mi) {
    const _Float16* p = qln16 + (size_t)(m0 + mw + mi * 16 + ml) * DC + kh;
    af[mi][0] = cat8(*(const v8h*)(p),      *(const v8h*)(p + 16));
    af[mi][1] = cat8(*(const v8h*)(p + 32), *(const v8h*)(p + 48));
  }
  int nl = lane & 15, k0f = (lane >> 4) * 8;
  for (int ni = 0; ni < 2; ++ni) {
    const _Float16* q = WpT + (size_t)(n0 + nw + ni * 16 + nl) * DC + k0f;
    v16h b0 = cat8(*(const v8h*)(q),      *(const v8h*)(q + 16));
    v16h b1 = cat8(*(const v8h*)(q + 32), *(const v8h*)(q + 48));
    for (int mi = 0; mi < 2; ++mi) {
      acc[mi][ni] = wmma16(af[mi][0], b0, acc[mi][ni]);
      acc[mi][ni] = wmma16(af[mi][1], b1, acc[mi][ni]);
    }
  }
  int mh = (lane >> 4) * 8;
  for (int mi = 0; mi < 2; ++mi)
    for (int ni = 0; ni < 2; ++ni) {
      int n = nw + ni * 16 + nl;
      float bb = bp[n0 + n];
      for (int i = 0; i < 8; ++i)
        Os[(mw + mi * 16 + mh + i) * 64 + n] = acc[mi][ni][i] + bb;
    }
  __syncthreads();
  for (int i = 0; i < 8; ++i) {
    int ch = t + i * 256;                         // 2048 chunks of 4 floats
    int r = ch >> 3, c4 = (ch & 7) * 4;
    *(float4*)&out[(size_t)(m0 + r) * DM + n0 + c4] = *(const float4*)&Os[r * 64 + c4];
  }
}

// ---------------- launch ----------------
extern "C" void kernel_launch(void* const* d_in, const int* in_sizes, int n_in,
                              void* d_out, int out_size, void* d_ws, size_t ws_size,
                              hipStream_t stream) {
  const float* Z     = (const float*)d_in[0];
  const float* W1    = (const float*)d_in[1];
  const float* b1    = (const float*)d_in[2];
  const float* W2    = (const float*)d_in[3];
  const float* b2    = (const float*)d_in[4];
  const float* codex = (const float*)d_in[5];
  const float* lnw   = (const float*)d_in[6];
  const float* lnb   = (const float*)d_in[7];
  const float* Wp    = (const float*)d_in[8];
  const float* bp    = (const float*)d_in[9];

  float* out   = (float*)d_out;
  float* loss  = out + (size_t)NTOK * DM;            // offset 8388608
  float* probs = loss + 1;                           // offset 8388609

  char* ws = (char*)d_ws;
  constexpr size_t OFF_CXH  = 0;                         // 8192*64 f16
  constexpr size_t OFF_CN2  = OFF_CXH  + 1048576;        // 8192 f32
  constexpr size_t OFF_WPT  = OFF_CN2  + 32768;          // 1024*64 f16 (transposed)
  constexpr size_t OFF_H16  = OFF_WPT  + 131072;         // 8192*1024 f16
  constexpr size_t OFF_ZE   = OFF_H16  + 16777216;       // 8192*64 f32
  constexpr size_t OFF_ZE16 = OFF_ZE   + 2097152;        // 8192*64 f16
  constexpr size_t OFF_IDX  = OFF_ZE16 + 1048576;        // 8192 i32
  constexpr size_t OFF_QLN  = OFF_IDX  + 32768;          // 8192*64 f16
  constexpr size_t OFF_PART = OFF_QLN  + 1048576;        // 32 f32

  _Float16* cxh   = (_Float16*)(ws + OFF_CXH);
  float*    cn2   = (float*)   (ws + OFF_CN2);
  _Float16* WpT   = (_Float16*)(ws + OFF_WPT);
  _Float16* H16   = (_Float16*)(ws + OFF_H16);
  float*    Ze    = (float*)   (ws + OFF_ZE);
  _Float16* Ze16  = (_Float16*)(ws + OFF_ZE16);
  int*      idxo  = (int*)     (ws + OFF_IDX);
  _Float16* qln16 = (_Float16*)(ws + OFF_QLN);
  float*    part  = (float*)   (ws + OFF_PART);

  // prep
  k_prep_codex<<<(KC + 255) / 256, 256, 0, stream>>>(codex, cxh, cn2);
  k_twp<<<(DC * DM + 255) / 256, 256, 0, stream>>>(Wp, WpT);
  k_zero_probs<<<2048, 256, 0, stream>>>(probs);

  // pipeline
  dim3 g1(DM / 64, NTOK / 128);
  k_gemm1<<<g1, 256, 0, stream>>>(Z, W1, b1, H16);
  k_gemm2<<<NTOK / 128, 256, 0, stream>>>(H16, W2, b2, Ze, Ze16);
  k_argmax<<<NTOK / 128, 256, 0, stream>>>(Ze16, cxh, cn2, idxo, probs);
  k_vq_ln<<<NTOK / 256, 256, 0, stream>>>(Ze, codex, idxo, lnw, lnb, qln16, part);
  k_loss_final<<<1, 32, 0, stream>>>(part, loss);
  dim3 g3(DM / 64, NTOK / 128);
  k_gemm3<<<g3, 256, 0, stream>>>(qln16, WpT, bp, out);
}
